// HashGridwithMLP_3530463117369
// MI455X (gfx1250) — compile-verified, matrix operand order checked by reading, not run on hardware
//
#include <hip/hip_runtime.h>
#include <math.h>

typedef __attribute__((ext_vector_type(16))) _Float16 v16h;
typedef __attribute__((ext_vector_type(8)))  _Float16 v8h;
typedef __attribute__((ext_vector_type(2)))  _Float16 v2h;
typedef __attribute__((ext_vector_type(8)))  float    v8f;

#define BM        128                 // rows (points) per block
#define SA0       40                  // layer0 act stride (halves): 32 feat cols + 8 pad
#define SW1       40                  // W1^T (first 32 rows only) k-stride
#define SAB       136                 // 128-col activation stride + 8 pad
#define SW2       136
#define SW3       136
#define W1T_OFF   0
#define W2T_OFF   (128 * SW1)                 // 5120 halves
#define W3T_OFF   (W2T_OFF + 128 * SW2)       // 22528 halves
#define LDSW_HALVES W3T_OFF                   // W1^T + W2^T staged in LDS (45056 B)
#define WS_W_HALVES (W3T_OFF + 48 * SW3)      // 29056 halves = 58112 bytes of f16 weights
#define BIAS1_BYTE_OFF  (WS_W_HALVES * 2)     // 58112: 128 f32 effective bias1
#define LOSS_BYTE_OFF   (BIAS1_BYTE_OFF + 512)
#define OUT3_STRIDE 48                        // layer3 out cols padded 42 -> 48 (f32)

struct ResLevels { float r[16]; };

// 16-bit A/B WMMA fragment for 16x16x32: caller pre-offsets p by kb
// (0 for lanes 0-15, 8 for lanes 16-31). frag[0..7] = K kb+0..7, frag[8..15] = K kb+16..23.
__device__ __forceinline__ v16h load_frag16(const _Float16* p) {
  v8h lo = *(const v8h*)(p);
  v8h hi = *(const v8h*)(p + 16);
  v16h r;
#pragma unroll
  for (int i = 0; i < 8; ++i) { r[i] = lo[i]; r[i + 8] = hi[i]; }
  return r;
}

// One MLP layer on a 128-row tile: wave owns row-tile `wave`, loops NTILES
// column tiles, K = KTILES*32. Bias + optional ReLU fused into the store.
template <int KTILES, int NTILES, bool RELU, bool OUTF>
__device__ __forceinline__ void gemm_layer(
    const _Float16* actIn, int sIn,
    const _Float16* wT,    int sW,
    const float* bias, int nBias,
    _Float16* outH, float* outF, int sOut,
    int wave, int lane)
{
  const int n     = lane & 15;
  const int kb    = (lane < 16) ? 0 : 8;   // K sub-offset per ISA 16-bit A/B layout
  const int mbase = (lane < 16) ? 0 : 8;   // C/D: lanes 16-31 hold rows M=8..15
  const _Float16* aRow = actIn + (wave * 16 + n) * sIn + kb;

  v8f acc[NTILES] = {};

#pragma unroll
  for (int kt = 0; kt < KTILES; ++kt) {
    const int k0 = kt * 32;
    const v16h a = load_frag16(aRow + k0);
#pragma unroll
    for (int c = 0; c < NTILES; ++c) {
      const v16h b = load_frag16(wT + (c * 16 + n) * sW + k0 + kb);
      acc[c] = __builtin_amdgcn_wmma_f32_16x16x32_f16(
          false, a, false, b, (short)0, acc[c], false, false);
    }
  }

#pragma unroll
  for (int c = 0; c < NTILES; ++c) {
    const int col = c * 16 + n;
    const float bv = (col < nBias) ? bias[col] : 0.0f;
#pragma unroll
    for (int r = 0; r < 8; ++r) {
      float v = acc[c][r] + bv;
      if (RELU) v = fmaxf(v, 0.0f);
      const int orow = wave * 16 + mbase + r;
      if (OUTF) outF[orow * sOut + col] = v;
      else      outH[orow * sOut + col] = (_Float16)v;
    }
  }
}

// Convert weights to f16, transposed K-major per output column, padded strides.
// Only W1 rows 0..31 (the hashgrid part) are kept; the cond part is folded into
// an effective fp32 bias: bias1eff = b1 + cond @ W1[32:,:]. W3 cols padded 42->48.
__global__ __launch_bounds__(256) void convert_weights(
    const float* __restrict__ W1, const float* __restrict__ W2,
    const float* __restrict__ W3, const float* __restrict__ cond,
    const float* __restrict__ b1,
    _Float16* __restrict__ ws, float* __restrict__ bias1eff)
{
  const int stride = gridDim.x * blockDim.x;
  const int t0 = blockIdx.x * blockDim.x + threadIdx.x;
  for (int i = t0; i < 32 * 128; i += stride) {            // W1 rows k<32
    const int k = i >> 7, n = i & 127;
    ws[W1T_OFF + n * SW1 + k] = (_Float16)W1[i];
  }
  for (int i = t0; i < 128 * 128; i += stride) {           // W2 (k<128, n<128)
    const int k = i >> 7, n = i & 127;
    ws[W2T_OFF + n * SW2 + k] = (_Float16)W2[i];
  }
  for (int i = t0; i < 48 * 128; i += stride) {            // W3 dest-indexed (n<48, k<128)
    const int n = i >> 7, k = i & 127;
    ws[W3T_OFF + n * SW3 + k] = (n < 42) ? (_Float16)W3[k * 42 + n] : (_Float16)0.0f;
  }
  if (blockIdx.x == 0 && threadIdx.x < 128) {              // fold cond@W1[32:,:] + b1 (fp32)
    const int n = threadIdx.x;
    float s = b1[n];
    for (int k = 0; k < 128; ++k) s += cond[k] * W1[(32 + k) * 128 + n];
    bias1eff[n] = s;
  }
}

__global__ __launch_bounds__(256) void fused_kernel(
    const float* __restrict__ xyz, const float* __restrict__ scaling,
    const float* __restrict__ rotation,
    const float* __restrict__ table,
    const float* __restrict__ aabb_min, const float* __restrict__ aabb_max,
    const float* __restrict__ bias1eff, const float* __restrict__ b2,
    const float* __restrict__ b3,
    const _Float16* __restrict__ wsW,
    float* __restrict__ lossPartial, float* __restrict__ out,
    int N, unsigned tmask, ResLevels rl)
{
  __shared__ __align__(16) _Float16 ldsW[LDSW_HALVES];   // 45056 B: W1^T + W2^T
  __shared__ __align__(16) _Float16 actA[BM * SAB];      // 34816 B (layer0-in / layer2-out)
  __shared__ __align__(16) _Float16 actB[BM * SAB];      // 34816 B (layer1-out / f32 out3)

  const int tid  = threadIdx.x;
  const int wave = tid >> 5;
  const int lane = tid & 31;

  // ---- stage W1^T / W2^T into LDS (contiguous b128 copies) ----
  {
    const uint4* src = (const uint4*)wsW;
    uint4* dst = (uint4*)ldsW;
    for (int i = tid; i < (LDSW_HALVES * 2) / 16; i += 256) dst[i] = src[i];
  }

  // ---- hashgrid encode -> actA[:, 0:32] (2 threads/row x 8 levels each) ----
  {
    const int row   = tid & 127;
    const int lbase = (tid >> 7) * 8;        // wave-uniform level half
    int gr = blockIdx.x * BM + row;
    if (gr > N - 1) gr = N - 1;              // tail rows compute garbage, masked later
    const float mn0 = aabb_min[0], mn1 = aabb_min[1], mn2 = aabb_min[2];
    const float mx0 = aabb_max[0], mx1 = aabb_max[1], mx2 = aabb_max[2];
    const float hi = 1.0f - 1e-6f;
    float x0 = fminf(fmaxf((xyz[gr * 3 + 0] - mn0) / (mx0 - mn0), 0.0f), hi);
    float y0 = fminf(fmaxf((xyz[gr * 3 + 1] - mn1) / (mx1 - mn1), 0.0f), hi);
    float z0 = fminf(fmaxf((xyz[gr * 3 + 2] - mn2) / (mx2 - mn2), 0.0f), hi);
    const size_t lstride = (size_t)(tmask + 1) * 2;
#pragma unroll
    for (int l = lbase; l < lbase + 8; ++l) {
      const float res = rl.r[l];
      const float xs = x0 * res, ys = y0 * res, zs = z0 * res;
      const float fx = floorf(xs), fy = floorf(ys), fz = floorf(zs);
      const float tx = xs - fx, ty = ys - fy, tz = zs - fz;
      const unsigned cx = (unsigned)fx, cy = (unsigned)fy, cz = (unsigned)fz;
      const float* tb = table + (size_t)l * lstride;
      float f0 = 0.0f, f1 = 0.0f;
#pragma unroll
      for (int c = 0; c < 8; ++c) {
        const unsigned hx = cx + ((c >> 2) & 1);
        const unsigned hy = cy + ((c >> 1) & 1);
        const unsigned hz = cz + (c & 1);
        const unsigned idx = (hx ^ (hy * 2654435761u) ^ (hz * 805459861u)) & tmask;
        const float w = ((c & 4) ? tx : 1.0f - tx) *
                        ((c & 2) ? ty : 1.0f - ty) *
                        ((c & 1) ? tz : 1.0f - tz);
        const float2 e = *(const float2*)(tb + (size_t)idx * 2);
        f0 += w * e.x; f1 += w * e.y;
      }
      v2h pk; pk[0] = (_Float16)f0; pk[1] = (_Float16)f1;
      *(v2h*)&actA[row * SA0 + 2 * l] = pk;
    }
  }
  __syncthreads();

  // ---- MLP: 3 WMMA layers (cond folded into bias1eff), LDS ping-pong ----
  gemm_layer<1, 8, true,  false>(actA, SA0, ldsW + W1T_OFF, SW1, bias1eff, 128,
                                 actB, nullptr, SAB, wave, lane);
  __syncthreads();
  gemm_layer<4, 8, true,  false>(actB, SAB, ldsW + W2T_OFF, SW2, b2, 128,
                                 actA, nullptr, SAB, wave, lane);
  __syncthreads();
  gemm_layer<4, 3, false, true >(actA, SAB, wsW + W3T_OFF, SW3, b3, 42,
                                 nullptr, (float*)actB, OUT3_STRIDE, wave, lane);
  __syncthreads();

  // ---- epilogue: base adds, output row, per-row losses ----
  const float* out3 = (const float*)actB;
  float* rowLoss = (float*)actA;     // actA dead; reuse as f32 scratch
  if (tid < BM) {
    const int gr = blockIdx.x * BM + tid;
    float l0 = 0.0f, l1 = 0.0f, l2 = 0.0f;
    if (gr < N) {
      const float* d = out3 + tid * OUT3_STRIDE;
      float* o = out + (size_t)gr * 42;
      const float d0 = d[0], d1 = d[1], d2 = d[2];
      o[0] = xyz[gr * 3 + 0] + d0;
      o[1] = xyz[gr * 3 + 1] + d1;
      o[2] = xyz[gr * 3 + 2] + d2;
#pragma unroll
      for (int j = 0; j < 3; ++j) { const float v = d[3 + j]; o[3 + j] = scaling[gr * 3 + j] + v; l1 += fabsf(v); }
#pragma unroll
      for (int j = 0; j < 4; ++j) { const float v = d[6 + j]; o[6 + j] = rotation[gr * 4 + j] + v; l2 += fabsf(v); }
#pragma unroll
      for (int j = 10; j < 42; ++j) o[j] = d[j];
      l0 = sqrtf(d0 * d0 + d1 * d1 + d2 * d2);
    }
    rowLoss[tid * 4 + 0] = l0;
    rowLoss[tid * 4 + 1] = l1;
    rowLoss[tid * 4 + 2] = l2;
  }
  __syncthreads();
#pragma unroll
  for (int s = 64; s > 0; s >>= 1) {       // fixed-order tree -> deterministic
    if (tid < s) {
      rowLoss[tid * 4 + 0] += rowLoss[(tid + s) * 4 + 0];
      rowLoss[tid * 4 + 1] += rowLoss[(tid + s) * 4 + 1];
      rowLoss[tid * 4 + 2] += rowLoss[(tid + s) * 4 + 2];
    }
    __syncthreads();
  }
  if (tid == 0) {
    lossPartial[blockIdx.x * 4 + 0] = rowLoss[0];
    lossPartial[blockIdx.x * 4 + 1] = rowLoss[1];
    lossPartial[blockIdx.x * 4 + 2] = rowLoss[2];
  }
}

__global__ __launch_bounds__(256) void finalize_kernel(
    const float* __restrict__ lossPartial, int nblocks,
    float* __restrict__ out, int N)
{
  __shared__ float s0[256], s1[256], s2[256];
  const int t = threadIdx.x;
  float a0 = 0.0f, a1 = 0.0f, a2 = 0.0f;
  for (int i = t; i < nblocks; i += 256) {
    a0 += lossPartial[i * 4 + 0];
    a1 += lossPartial[i * 4 + 1];
    a2 += lossPartial[i * 4 + 2];
  }
  s0[t] = a0; s1[t] = a1; s2[t] = a2;
  __syncthreads();
  for (int s = 128; s > 0; s >>= 1) {
    if (t < s) { s0[t] += s0[t + s]; s1[t] += s1[t + s]; s2[t] += s2[t + s]; }
    __syncthreads();
  }
  float* tail = out + (size_t)N * 42;
  if (t == 0) {
    const float inv = 1.0f / (float)N;
    tail[0] = s0[0] * inv; tail[1] = s1[0] * inv; tail[2] = s2[0] * inv;
  }
  if (t >= 3 && t < 42) tail[t] = 0.0f;
}

extern "C" void kernel_launch(void* const* d_in, const int* in_sizes, int n_in,
                              void* d_out, int out_size, void* d_ws, size_t ws_size,
                              hipStream_t stream) {
  (void)n_in; (void)out_size; (void)ws_size;
  const float* xyz      = (const float*)d_in[0];
  const float* scaling  = (const float*)d_in[1];
  const float* rotation = (const float*)d_in[2];
  const float* cond     = (const float*)d_in[3];
  const float* table    = (const float*)d_in[4];
  const float* aabb_min = (const float*)d_in[5];
  const float* aabb_max = (const float*)d_in[6];
  const float* W1 = (const float*)d_in[7];
  const float* b1 = (const float*)d_in[8];
  const float* W2 = (const float*)d_in[9];
  const float* b2 = (const float*)d_in[10];
  const float* W3 = (const float*)d_in[11];
  const float* b3 = (const float*)d_in[12];

  const int N = in_sizes[0] / 3;
  const int nblocks = (N + BM - 1) / BM;
  const unsigned T = (unsigned)(in_sizes[4] / (16 * 2));   // 2^19
  const unsigned tmask = T - 1u;

  _Float16* wsW = (_Float16*)d_ws;
  float* bias1eff    = (float*)((char*)d_ws + BIAS1_BYTE_OFF);
  float* lossPartial = (float*)((char*)d_ws + LOSS_BYTE_OFF);
  float* out = (float*)d_out;

  ResLevels rl;                                 // fp64 host math mirrors numpy
  const double growth = pow(2048.0 / 16.0, 1.0 / 15.0);
  for (int l = 0; l < 16; ++l) rl.r[l] = (float)floor(16.0 * pow(growth, (double)l));

  convert_weights<<<96, 256, 0, stream>>>(W1, W2, W3, cond, b1, wsW, bias1eff);
  fused_kernel<<<nblocks, 256, 0, stream>>>(xyz, scaling, rotation, table,
                                            aabb_min, aabb_max, bias1eff, b2, b3,
                                            wsW, lossPartial, out, N, tmask, rl);
  finalize_kernel<<<1, 256, 0, stream>>>(lossPartial, nblocks, out, N);
}
